// Joiner_19610820674056
// MI455X (gfx1250) — matrix operand order repaired and checked by reading, not compiled, not run on hardware
//
#include <hip/hip_runtime.h>

typedef __attribute__((ext_vector_type(16))) _Float16 v16h;
typedef __attribute__((ext_vector_type(8)))  _Float16 v8h;
typedef __attribute__((ext_vector_type(4)))  _Float16 v4h;
typedef __attribute__((ext_vector_type(8)))  float    v8f;

// Problem constants (from reference): B=4, T=256, U=64, C=512, V=1024
constexpr int Bc = 4;
constexpr int Tc = 256;
constexpr int Uc = 64;
constexpr int Cc = 512;
constexpr int Vc = 1024;

constexpr int BN = 128;        // V columns per N-pass inside the workgroup

union Frag16 { v16h v; v8h h[2]; };

// Branch-free tanh: prefer the gfx1250 V_TANH_F32 instruction.
__device__ __forceinline__ float fast_tanh(float x) {
#if __has_builtin(__builtin_amdgcn_tanhf)
    return __builtin_amdgcn_tanhf(x);
#elif __has_builtin(__builtin_amdgcn_tanh_f32)
    return __builtin_amdgcn_tanh_f32(x);
#else
    // tanh(x) = 1 - 2/(exp(2x)+1); saturates to +/-1 without branches.
#if __has_builtin(__builtin_amdgcn_exp2f)
    float e = __builtin_amdgcn_exp2f(x * 2.8853900817779268f); // exp(2x)
#else
    float e = __expf(2.0f * x);
#endif
#if __has_builtin(__builtin_amdgcn_rcpf)
    return 1.0f - 2.0f * __builtin_amdgcn_rcpf(e + 1.0f);
#else
    return 1.0f - 2.0f / (e + 1.0f);
#endif
#endif
}

// XOR-swizzled LDS layout for the A tile: 64 rows x 512 halves = 65536 B.
// 16B granules swizzled by row -> conflict-light ds_load_b128, zero padding.
__device__ __forceinline__ int sw_idx(int u, int k) {
    return u * Cc + ((((k >> 3) ^ (u & 7)) << 3) | (k & 7));
}

// ---------------------------------------------------------------------------
// Kernel 1: W (V x C, f32, row-major) -> Wt (C x V, f16), LDS-tiled 32x32
// transpose: coalesced reads AND writes.  Wt (1 MB) then lives in L2.
// ---------------------------------------------------------------------------
__global__ __launch_bounds__(256)
void wt_convert_kernel(const float* __restrict__ W, _Float16* __restrict__ Wt) {
    __shared__ _Float16 tile[32][34];       // odd-ish stride: conflict-free cols

    const int v0 = blockIdx.x * 32;         // V tile origin
    const int c0 = blockIdx.y * 32;         // C tile origin
    const int tx = threadIdx.x & 31;
    const int ty = threadIdx.x >> 5;        // 0..7

    #pragma unroll
    for (int r = 0; r < 32; r += 8) {       // coalesced read along C
        const int v = v0 + ty + r;
        tile[ty + r][tx] = (_Float16)W[(size_t)v * Cc + c0 + tx];
    }
    __syncthreads();
    #pragma unroll
    for (int r = 0; r < 32; r += 8) {       // coalesced write along V
        const int c = c0 + ty + r;
        Wt[(size_t)c * Vc + v0 + tx] = tile[tx][ty + r];
    }
}

// ---------------------------------------------------------------------------
// Kernel 2: one workgroup per (b,t).  Stage A[u][c] = f16(tanh(enc+dec))
// ONCE (full K) in swizzled LDS, then sweep all 8 N-tiles of 128 columns,
// each a 64x128x512 WMMA GEMM vs Wt, + bias, f32 stores.
// Block: 256 threads = 8 waves, tiled 2(M) x 4(N); 2x2 16x16 accs per wave.
// ---------------------------------------------------------------------------
__global__ __launch_bounds__(256)
void joiner_wmma_kernel(const float* __restrict__ enc,
                        const float* __restrict__ dec,
                        const _Float16* __restrict__ Wt,
                        const float* __restrict__ bias,
                        float* __restrict__ out) {
    __shared__ _Float16 sA[Uc * Cc];            // exactly 65536 B

    const int tid  = threadIdx.x;
    const int lane = tid & 31;                  // wave32
    const int wave = tid >> 5;
    const int wm   = wave >> 2;                 // 0..1  -> 32 M-rows each
    const int wn   = wave & 3;                  // 0..3  -> 32 N-cols each

    const int bt   = blockIdx.x;                // b*T + t
    const int bidx = bt >> 8;                   // / T (T=256)

    const float* encRow  = enc + (size_t)bt * Cc;
    const float* decBase = dec + (size_t)bidx * Uc * Cc;

    const int mBase = wm * 32;

    // ---- stage full A tile: 64x512 elements, float4-vectorized ----
    #pragma unroll
    for (int i = 0; i < 32; ++i) {
        int e  = i * 256 + tid;                 // 0..8191
        int u  = e >> 7;                        // 0..63
        int c4 = (e & 127) << 2;                // 0..508, step 4
        const float4 ev = *(const float4*)(encRow + c4);
        const float4 dv = *(const float4*)(decBase + (size_t)u * Cc + c4);
        v4h hv;
        hv[0] = (_Float16)fast_tanh(ev.x + dv.x);
        hv[1] = (_Float16)fast_tanh(ev.y + dv.y);
        hv[2] = (_Float16)fast_tanh(ev.z + dv.z);
        hv[3] = (_Float16)fast_tanh(ev.w + dv.w);
        *(v4h*)&sA[sw_idx(u, c4)] = hv;         // 8B DS store, aligned
    }
    __syncthreads();

    const int kfirst = (lane >> 4) << 3;        // A frag: 0 (lanes<16) or 8
    const int mOff   = (lane >> 4) * 8;         // C/D frag row offset

    // ---- sweep the 8 N-tiles, reusing the staged A tile ----
    for (int n0 = 0; n0 < Vc; n0 += BN) {
        const int nBase = n0 + wn * 32;

        v8f acc[2][2] = {};

        #pragma unroll
        for (int kk = 0; kk < Cc; kk += 32) {
            // B fragments (32x16, K x N): lane%16 = N col, K {kfirst..+7, kfirst+16..+23}
            const _Float16* wrow = Wt + (size_t)(kk + lane) * Vc;
            Frag16 bf[2];
            #pragma unroll
            for (int ni = 0; ni < 2; ++ni) {
                const _Float16* wp = Wt + (size_t)kk * Vc + nBase + ni * 16 + (lane & 15);
                #pragma unroll
                for (int s = 0; s < 8; ++s) {
                    bf[ni].v[s]     = wp[(size_t)(kfirst + s) * Vc];
                    bf[ni].v[s + 8] = wp[(size_t)(kfirst + 16 + s) * Vc];
                }
            }
            if (kk + 32 < Cc)                   // WGP-scope prefetch, next K rows
                __builtin_prefetch(wrow + (size_t)32 * Vc + nBase, 0, 3);

            // A fragments (16x32): lanes 0-15 hold K {0..7,16..23},
            // lanes 16-31 hold K {8..15,24..31}
            Frag16 af[2];
            #pragma unroll
            for (int mi = 0; mi < 2; ++mi) {
                const int r = mBase + mi * 16 + (lane & 15);
                af[mi].h[0] = *(const v8h*)&sA[sw_idx(r, kk + kfirst)];
                af[mi].h[1] = *(const v8h*)&sA[sw_idx(r, kk + kfirst + 16)];
            }

            #pragma unroll
            for (int mi = 0; mi < 2; ++mi)
                #pragma unroll
                for (int ni = 0; ni < 2; ++ni)
                    acc[mi][ni] = __builtin_amdgcn_wmma_f32_16x16x32_f16(
                        /*neg_a=*/false, af[mi].v,
                        /*neg_b=*/false, bf[ni].v,
                        /*c_mod=*/(short)0, acc[mi][ni],
                        /*reuse_a=*/false, /*reuse_b=*/false);
        }

        // ---- epilogue: lanes 0-15 -> M = vgpr, lanes 16-31 -> M = 8+vgpr
        #pragma unroll
        for (int ni = 0; ni < 2; ++ni) {
            const int vcol = nBase + ni * 16 + (lane & 15);
            const float bv = bias[vcol];
            #pragma unroll
            for (int mi = 0; mi < 2; ++mi) {
                const int uLo = mBase + mi * 16 + mOff;
                float* op = out + ((size_t)bt * Uc + uLo) * Vc + vcol;
                #pragma unroll
                for (int r = 0; r < 8; ++r)
                    op[(size_t)r * Vc] = acc[mi][ni][r] + bv;
            }
        }
    }
}

// ---------------------------------------------------------------------------
extern "C" void kernel_launch(void* const* d_in, const int* in_sizes, int n_in,
                              void* d_out, int out_size, void* d_ws, size_t ws_size,
                              hipStream_t stream) {
    const float* enc  = (const float*)d_in[0];   // (B,T,C)
    const float* dec  = (const float*)d_in[1];   // (B,U,C)
    const float* W    = (const float*)d_in[2];   // (V,C)
    const float* bias = (const float*)d_in[3];   // (V,)
    float*       out  = (float*)d_out;           // (B,T,U,V)
    _Float16*    Wt   = (_Float16*)d_ws;         // (C,V) f16, 1 MB

    // 1) tiled transpose+convert W -> f16
    dim3 tgrid(Vc / 32, Cc / 32);
    wt_convert_kernel<<<tgrid, 256, 0, stream>>>(W, Wt);

    // 2) fused joiner GEMM: one WG per (b,t)
    joiner_wmma_kernel<<<Bc * Tc, 256, 0, stream>>>(enc, dec, Wt, bias, out);
}
